// Head_45045617000872
// MI455X (gfx1250) — matrix operand order repaired and checked by reading, not compiled, not run on hardware
//
#include <hip/hip_runtime.h>

// ---------------------------------------------------------------------------
// Causal single-head attention, B=8 T=4096 C=768 H=64 (fp32 in/out).
// Compute-bound -> bf16 WMMA (V_WMMA_F32_16X16X32_BF16), fp32 accumulation.
//  - q pre-scaled by 768^-0.5 ; softmax denominator via WMMA (P @ ones).
//  - proj K-loop fully unrolled (24 chunks): no backedge phis -> no register
//    copies, scheduler hoists load clauses freely across chunks.
//  - per-wave LDS tile for the P (C->A layout) round trip; LDS is in-order
//    within a wave, so compiler-only memory barriers replace s_barrier.
// ---------------------------------------------------------------------------

#define DEV __device__ __forceinline__
#define MEMBAR() asm volatile("" ::: "memory")

typedef __attribute__((ext_vector_type(16))) __bf16        bf16x16;
typedef __attribute__((ext_vector_type(8)))  float         f32x8;
typedef __attribute__((ext_vector_type(4)))  float         f32x4;
typedef __attribute__((ext_vector_type(4)))  unsigned int  u32x4;

union FragU {
    u32x4          q[2];
    bf16x16        f;
    unsigned short s[16];
};

DEV unsigned short bf16_bits(float f) {
    union { __bf16 h; unsigned short s; } u;
    u.h = (__bf16)f;               // native v_cvt, RNE
    return u.s;
}

// 16 contiguous bf16 (B-fragment lane slice: K = 16*half .. +15 of column n)
DEV bf16x16 load_frag16(const unsigned short* p) {
    FragU u;
    u.q[0] = *(const u32x4*)(p);
    u.q[1] = *(const u32x4*)(p + 8);
    return u.f;
}

// A-fragment lane slice: two runs of 8 bf16 (used adjacent: K = 16*half .. +15 of row m)
DEV bf16x16 load_frag_split(const unsigned short* p0, const unsigned short* p1) {
    FragU u;
    u.q[0] = *(const u32x4*)(p0);
    u.q[1] = *(const u32x4*)(p1);
    return u.f;
}

DEV f32x8 wmma_bf16(bf16x16 a, bf16x16 b, f32x8 c) {
    return __builtin_amdgcn_wmma_f32_16x16x32_bf16(
        false, a, false, b, (short)0, c, false, false);
}

// ---------------------------------------------------------------------------
// Kernel 0: W[768][64] fp32 -> WT[64][768] bf16 for q,k,v
// ---------------------------------------------------------------------------
__global__ void prep_wt(const float* __restrict__ Wk, const float* __restrict__ Wq,
                        const float* __restrict__ Wv, unsigned short* __restrict__ WT) {
    int idx = blockIdx.x * 256 + threadIdx.x;
    if (idx >= 3 * 64 * 768) return;
    int w   = idx / (64 * 768);
    int rem = idx - w * (64 * 768);
    int h   = rem / 768;
    int c   = rem - h * 768;
    const float* W = (w == 0) ? Wq : ((w == 1) ? Wk : Wv);
    WT[idx] = bf16_bits(W[c * 64 + h]);
}

// ---------------------------------------------------------------------------
// Kernel 1: one projection (w = bid%3: 0=q,1=k,2=v) for a 64-row tile.
// 4 waves/block, 16 rows/wave, 96 WMMAs/wave. K-loop fully unrolled.
// Adjacent blocks share the same x rows -> 2nd/3rd reads hit L2.
// ---------------------------------------------------------------------------
__global__ __launch_bounds__(128)
void proj_one(const float* __restrict__ x, const unsigned short* __restrict__ WT,
              unsigned short* __restrict__ qo, unsigned short* __restrict__ ko,
              unsigned short* __restrict__ vto) {
    const int bid   = blockIdx.x;
    const int w     = bid % 3;
    const int rt    = bid / 3;
    const int wave  = threadIdx.x >> 5;
    const int lane  = threadIdx.x & 31;
    const int half  = lane >> 4;
    const int ln    = lane & 15;
    const int rbase = rt * 64 + wave * 16;

    const float*          xr = x + (size_t)(rbase + ln) * 768;
    const unsigned short* wb = WT + (size_t)w * 49152 + (size_t)ln * 768 + 16 * half;

    const f32x8 vz = {};
    f32x8 acc[4];
    #pragma unroll
    for (int ch = 0; ch < 4; ++ch) acc[ch] = vz;

    #pragma unroll
    for (int c = 0; c < 24; ++c) {
        const float*          xc = xr + c * 32;
        const unsigned short* pc = wb + c * 32;
        f32x4 a0 = *(const f32x4*)(xc + 16 * half);
        f32x4 a1 = *(const f32x4*)(xc + 16 * half + 4);
        f32x4 a2 = *(const f32x4*)(xc + 16 * half + 8);
        f32x4 a3 = *(const f32x4*)(xc + 16 * half + 12);
        bf16x16 bfr[4];
        #pragma unroll
        for (int ch = 0; ch < 4; ++ch) bfr[ch] = load_frag16(pc + ch * 12288);

        bf16x16 av;
        #pragma unroll
        for (int i = 0; i < 4; ++i) {
            av[i]      = (__bf16)a0[i];
            av[4 + i]  = (__bf16)a1[i];
            av[8 + i]  = (__bf16)a2[i];
            av[12 + i] = (__bf16)a3[i];
        }
        #pragma unroll
        for (int ch = 0; ch < 4; ++ch) acc[ch] = wmma_bf16(av, bfr[ch], acc[ch]);
    }

    // Epilogue: branch on w once, then unrolled stores.
    const float qscale = 0.036084391824351613f;  // 768^-0.5 folded into q
    if (w == 0) {
        #pragma unroll
        for (int ch = 0; ch < 4; ++ch) {
            int h = 16 * ch + ln;
            #pragma unroll
            for (int vv = 0; vv < 8; ++vv) {
                size_t row = (size_t)rbase + vv + 8 * half;
                qo[row * 64 + h] = bf16_bits(acc[ch][vv] * qscale);
            }
        }
    } else if (w == 1) {
        #pragma unroll
        for (int ch = 0; ch < 4; ++ch) {
            int h = 16 * ch + ln;
            #pragma unroll
            for (int vv = 0; vv < 8; ++vv) {
                size_t row = (size_t)rbase + vv + 8 * half;
                ko[row * 64 + h] = bf16_bits(acc[ch][vv]);
            }
        }
    } else {
        #pragma unroll
        for (int ch = 0; ch < 4; ++ch) {
            int h = 16 * ch + ln;
            #pragma unroll
            for (int vv = 0; vv < 8; ++vv) {
                size_t row = (size_t)rbase + vv + 8 * half;
                int bb = (int)(row >> 12);
                int t  = (int)(row & 4095);
                vto[((size_t)bb * 64 + h) * 4096 + t] = bf16_bits(acc[ch][vv]);
            }
        }
    }
}

// ---------------------------------------------------------------------------
// Kernel 2: flash attention forward. 4 waves/block, 16 query rows/wave,
// 32-key blocks, per-wave trip count, ping-pong K-fragment prefetch.
// ---------------------------------------------------------------------------
__global__ __launch_bounds__(128)
void attn_fwd(const unsigned short* __restrict__ q, const unsigned short* __restrict__ k,
              const unsigned short* __restrict__ vt, float* __restrict__ out) {
    __shared__ __align__(16) unsigned short ldsP[4][16][32];

    const int wave  = threadIdx.x >> 5;
    const int lane  = threadIdx.x & 31;
    const int half  = lane >> 4;
    const int ln    = lane & 15;
    const int b     = blockIdx.x >> 6;
    const int qwg   = (blockIdx.x & 63) * 64;
    const int qbase = qwg + wave * 16;

    // Q (pre-scaled) as two A fragments, row per lane = ln
    const unsigned short* qrow = q + ((size_t)b * 4096 + qbase + ln) * 64;
    bf16x16 aq0 = load_frag_split(qrow + 16 * half,      qrow + 16 * half + 8);
    bf16x16 aq1 = load_frag_split(qrow + 32 + 16 * half, qrow + 40 + 16 * half);

    FragU onesu;
    #pragma unroll
    for (int i = 0; i < 16; ++i) onesu.s[i] = 0x3F80;   // bf16(1.0)
    const bf16x16 bones = onesu.f;

    float mrow[8];
    #pragma unroll
    for (int i = 0; i < 8; ++i) mrow[i] = -__builtin_inff();
    const f32x8 vz = {};
    f32x8 acc[5];                       // [0..3] = O chunks, [4] = l (row sums)
    #pragma unroll
    for (int c = 0; c < 5; ++c) acc[c] = vz;

    const int    nblocks = (qbase >> 5) + 1;   // per-wave causal extent
    const size_t kbase   = (size_t)b * 4096 * 64;
    const size_t vtb     = (size_t)b * 64;

    auto load_kf = [&](int kb, bf16x16 (&kf)[4]) {
        const unsigned short* kr0 = k + kbase + (size_t)(kb + ln)      * 64;
        const unsigned short* kr1 = k + kbase + (size_t)(kb + 16 + ln) * 64;
        kf[0] = load_frag16(kr0 + 16 * half);
        kf[1] = load_frag16(kr0 + 32 + 16 * half);
        kf[2] = load_frag16(kr1 + 16 * half);
        kf[3] = load_frag16(kr1 + 32 + 16 * half);
    };

    auto do_block = [&](int blk, bf16x16 (&kfc)[4], bf16x16 (&kfn)[4]) {
        const int kb = blk * 32;

        // S = Q K^T : 16 rows x 32 keys as two 16x16 C tiles
        f32x8 s0 = vz, s1 = vz;
        s0 = wmma_bf16(aq0, kfc[0], s0);
        s0 = wmma_bf16(aq1, kfc[1], s0);
        s1 = wmma_bf16(aq0, kfc[2], s1);
        s1 = wmma_bf16(aq1, kfc[3], s1);

        // prefetch next block's K fragments into the other buffer
        const int kbn = (blk + 1 < nblocks) ? kb + 32 : kb;
        load_kf(kbn, kfn);
        // V fragments for this block; consumed after softmax -> latency hidden
        bf16x16 bv[4];
        #pragma unroll
        for (int c = 0; c < 4; ++c)
            bv[c] = load_frag16(vt + (vtb + 16 * c + ln) * 4096 + kb + 16 * half);
        __builtin_amdgcn_sched_barrier(0);   // keep prefetch above the math

        if (kb + 31 > qbase) {               // wave-uniform: diagonal blocks only
            #pragma unroll
            for (int vv = 0; vv < 8; ++vv) {
                const int row = qbase + vv + 8 * half;
                if (kb + ln      > row) s0[vv] = -__builtin_inff();
                if (kb + 16 + ln > row) s1[vv] = -__builtin_inff();
            }
        }

        // per-row tile max (reduce across the 16 lanes of each half-group)
        float tm[8];
        #pragma unroll
        for (int vv = 0; vv < 8; ++vv) tm[vv] = fmaxf(s0[vv], s1[vv]);
        #pragma unroll
        for (int st = 1; st < 16; st <<= 1)
            #pragma unroll
            for (int vv = 0; vv < 8; ++vv)
                tm[vv] = fmaxf(tm[vv], __shfl_xor(tm[vv], st, 16));

        // online rescale + P -> LDS (bf16, [row][key] for A-layout reload)
        float alpha[8];
        #pragma unroll
        for (int vv = 0; vv < 8; ++vv) {
            float mn  = fmaxf(mrow[vv], tm[vv]);
            alpha[vv] = __expf(mrow[vv] - mn);
            mrow[vv]  = mn;
            float p0  = __expf(s0[vv] - mn);
            float p1  = __expf(s1[vv] - mn);
            const int row = vv + 8 * half;
            ldsP[wave][row][ln]      = bf16_bits(p0);
            ldsP[wave][row][16 + ln] = bf16_bits(p1);
        }
        #pragma unroll
        for (int c = 0; c < 5; ++c)
            #pragma unroll
            for (int vv = 0; vv < 8; ++vv)
                acc[c][vv] *= alpha[vv];

        MEMBAR();   // order LDS stores before the A-layout reload (same wave)
        bf16x16 ap = load_frag_split(&ldsP[wave][ln][16 * half],
                                     &ldsP[wave][ln][16 * half + 8]);
        acc[4] = wmma_bf16(ap, bones, acc[4]);         // l += row-sum(P)
        #pragma unroll
        for (int c = 0; c < 4; ++c)
            acc[c] = wmma_bf16(ap, bv[c], acc[c]);     // O += P V
        MEMBAR();   // keep next iteration's stores after this read
    };

    bf16x16 kfA[4], kfB[4];
    load_kf(0, kfA);
    int blk = 0;
    for (; blk + 2 <= nblocks; blk += 2) {   // ping-pong, no register rotation
        do_block(blk,     kfA, kfB);
        do_block(blk + 1, kfB, kfA);
    }
    if (blk < nblocks) do_block(blk, kfA, kfB);

    // epilogue: out = O / l  (fp32, [B,T,H])
    float invl[8];
    #pragma unroll
    for (int vv = 0; vv < 8; ++vv) invl[vv] = 1.0f / acc[4][vv];
    #pragma unroll
    for (int c = 0; c < 4; ++c) {
        int h = 16 * c + ln;
        #pragma unroll
        for (int vv = 0; vv < 8; ++vv) {
            size_t row = (size_t)b * 4096 + qbase + vv + 8 * half;
            out[row * 64 + h] = acc[c][vv] * invl[vv];
        }
    }
}

// ---------------------------------------------------------------------------
// Host launcher
// ---------------------------------------------------------------------------
extern "C" void kernel_launch(void* const* d_in, const int* in_sizes, int n_in,
                              void* d_out, int out_size, void* d_ws, size_t ws_size,
                              hipStream_t stream) {
    const float* x  = (const float*)d_in[0];
    const float* Wk = (const float*)d_in[1];
    const float* Wq = (const float*)d_in[2];
    const float* Wv = (const float*)d_in[3];

    char* ws = (char*)d_ws;
    // Workspace: WT 294912 B | q 4 MB | k 4 MB | vT 4 MB  (~12.6 MB total)
    unsigned short* WT = (unsigned short*)(ws);
    unsigned short* qb = (unsigned short*)(ws + 294912);
    unsigned short* kb = (unsigned short*)(ws + 294912 + 4194304);
    unsigned short* vt = (unsigned short*)(ws + 294912 + 2 * 4194304);
    float* out = (float*)d_out;

    prep_wt <<<576, 256, 0, stream>>>(Wk, Wq, Wv, WT);
    proj_one<<<1536, 128, 0, stream>>>(x, WT, qb, kb, vt);
    attn_fwd<<<512, 128, 0, stream>>>(qb, kb, vt, out);
}
